// MetaworldSACMixtureMHActorNetwork_52536039964917
// MI455X (gfx1250) — matrix operand order repaired and checked by reading, not compiled, not run on hardware
//
#include <hip/hip_runtime.h>
#include <hip/hip_bf16.h>

// ---------------------------------------------------------------------------
// Fused MoE actor network for MI455X (gfx1250, wave32, WMMA).
//   Prep kernels transpose+convert all GEMM weights to bf16 [n][k] layout in
//   workspace, so B-fragments are two contiguous 16B loads. Main kernel fuses
//   backbone (2 layers x 8 experts), mixture combine, per-context heads.
//   One block = 32 batch rows (2 WMMA M-tiles), 512 threads (16 waves),
//   each wave owns 64 output columns. K-loops are t-outer / n-tile-inner so
//   each A fragment is loaded from LDS once and feeds 4 WMMAs (2 in heads). The mixture
//   accumulator lives in LDS (fp32); register working set stays < 256 VGPRs.
// ---------------------------------------------------------------------------

typedef __bf16 bf16_t;
typedef __attribute__((ext_vector_type(16))) __bf16 v16bf;
typedef __attribute__((ext_vector_type(8)))  __bf16 v8bf;
typedef __attribute__((ext_vector_type(4)))  __bf16 v4bf;
typedef __attribute__((ext_vector_type(8)))  float  v8f;

namespace {
constexpr int IN_D  = 39;     // obs dim (padded to 64 for K)
constexpr int KPAD1 = 64;     // padded K for layer 1
constexpr int H1    = 1024;
constexpr int H2    = 1024;
constexpr int HHD   = 512;
constexpr int OUTD  = 4;
constexpr int NE    = 8;      // experts
constexpr int NC    = 10;     // contexts
constexpr int BT    = 32;     // batch rows per block (2 M-tiles)
constexpr int THREADS = 512;  // 16 waves
constexpr int NT_BB = 4;      // backbone n-tiles per wave (64 cols)
constexpr int NT_H  = 2;      // head n-tiles per wave (32 cols)
constexpr int SSTR  = H1 + 8; // LDS row stride (bf16) for h / f tile
constexpr int FSTR  = H1 + 4; // LDS row stride (fp32) for f accumulator
constexpr int YSTR  = HHD + 8;

constexpr size_t W1T_ELEMS = (size_t)NE * H1 * KPAD1;   // [e][n][k<64]
constexpr size_t W2T_ELEMS = (size_t)NE * H2 * H1;      // [e][n][k]
constexpr size_t WHT_ELEMS = (size_t)NC * HHD * H2;     // [c][n][k]
}

__device__ __forceinline__ v8bf ld8(const bf16_t* p) { return *(const v8bf*)p; }

__device__ __forceinline__ v16bf mk_frag(v8bf lo, v8bf hi) {
    return __builtin_shufflevector(lo, hi, 0, 1, 2, 3, 4, 5, 6, 7,
                                           8, 9, 10, 11, 12, 13, 14, 15);
}

__device__ __forceinline__ v8f wmma_bf16(v16bf a, v16bf b, v8f c) {
    return __builtin_amdgcn_wmma_f32_16x16x32_bf16(false, a, false, b,
                                                   (short)0, c, false, false);
}

// ---------------------------------------------------------------------------
// Transpose + fp32->bf16 convert: in fp32 [S][K][N] -> out bf16 [S][N][Kpad],
// zero padding K..Kpad. grid = (S, Kpad/32, N/32), block = 256.
// ---------------------------------------------------------------------------
__global__ __launch_bounds__(256) void
transpose_cvt(const float* __restrict__ in, bf16_t* __restrict__ out,
              int K, int N, int Kpad)
{
    __shared__ bf16_t tile[32][33];
    const int s  = blockIdx.x, kt = blockIdx.y, nt = blockIdx.z;
    const int tid = threadIdx.x;
    const float* src = in + (size_t)s * K * N;
    bf16_t*      dst = out + (size_t)s * N * Kpad;

    const int nl = tid & 31;                 // coalesced over n
    #pragma unroll
    for (int r = 0; r < 4; ++r) {
        const int kl = (tid >> 5) * 4 + r;
        const int k = kt * 32 + kl, n = nt * 32 + nl;
        const float v = (k < K) ? src[(size_t)k * N + n] : 0.f;
        tile[kl][nl] = (bf16_t)v;
    }
    __syncthreads();
    const int kq  = tid & 7;                 // 4 contiguous k per thread
    const int nl2 = tid >> 3;
    v4bf w;
    #pragma unroll
    for (int r = 0; r < 4; ++r) w[r] = tile[kq * 4 + r][nl2];
    *(v4bf*)(dst + (size_t)(nt * 32 + nl2) * Kpad + kt * 32 + kq * 4) = w;
}

// ---------------------------------------------------------------------------
// Fused network. PREP=true: bf16 pre-transposed weights (fast path).
// ---------------------------------------------------------------------------
template <bool PREP>
__global__ __launch_bounds__(THREADS, 1) void
fused_moe_actor(const float* __restrict__ state,   // [B,39]
                const int*   __restrict__ cidx,    // [B]
                const float* __restrict__ Wte,     // [E,C]
                const float* __restrict__ Wb1,     // [E,39,1024]  (fp32 path)
                const float* __restrict__ bb1,     // [E,1024]
                const float* __restrict__ Wb2,     // [E,1024,1024](fp32 path)
                const float* __restrict__ bb2,     // [E,1024]
                const float* __restrict__ Wh1,     // [C,1024,512] (fp32 path)
                const float* __restrict__ bh1,     // [C,512]
                const float* __restrict__ Wh2,     // [C,512,4]
                const float* __restrict__ bh2,     // [C,4]
                const bf16_t* __restrict__ W1T,    // [E,1024,64]   bf16 n-major
                const bf16_t* __restrict__ W2T,    // [E,1024,1024] bf16 n-major
                const bf16_t* __restrict__ WhT,    // [C,512,1024]  bf16 n-major
                float* __restrict__ out)           // [B,4]
{
    __shared__ __align__(16) bf16_t sA[BT * KPAD1];  // state tile, K padded
    __shared__ __align__(16) bf16_t sH[BT * SSTR];   // h tile (reused for f)
    __shared__ __align__(16) float  sF[BT * FSTR];   // f accumulator (fp32)
    __shared__ __align__(16) bf16_t sY[BT * YSTR];   // head hidden tile
    __shared__ float sW[BT * NE];                    // mixture weights
    __shared__ int   sC[BT];                         // per-row context
    __shared__ int   sPresent[NC];

    const int tid  = threadIdx.x;
    const int lane = tid & 31;
    const int wave = tid >> 5;                       // 0..15
    const int b0   = blockIdx.x * BT;

    // ---- stage state tile (bf16, zero pad K 39->64) + contexts ----
    for (int i = tid; i < BT * KPAD1; i += THREADS) {
        const int m = i >> 6, k = i & 63;
        const float v = (k < IN_D) ? state[(b0 + m) * IN_D + k] : 0.f;
        sA[m * KPAD1 + k] = (bf16_t)v;
    }
    for (int i = tid; i < BT * FSTR; i += THREADS) sF[i] = 0.f;
    if (tid < NC) sPresent[tid] = 0;
    if (tid < BT) sC[tid] = cidx[b0 + tid];
    __syncthreads();
    if (tid < BT) sPresent[sC[tid]] = 1;
    if (tid < BT * NE) {                             // w[b,e] = W_te[e, c_b]
        const int m = tid >> 3, e = tid & 7;
        sW[m * NE + e] = Wte[e * NC + sC[m]];
    }

    // per-lane fragment geometry (16-bit WMMA layouts)
    const int nlo = lane & 15;                        // N (and A row) in tile
    const int ka  = (lane < 16) ? 0 : 8;              // A K-base
    const int kb  = (lane < 16) ? 0 : 8;              // B K-base
    const int mhi = (lane >> 4) * 8;                  // C/D row offset
    const int nbase  = wave * 64 + nlo;               // backbone column base
    const int nbaseH = wave * 32 + nlo;               // head column base

    __syncthreads();

    // =================== expert loop ===================
    for (int e = 0; e < NE; ++e) {
        // ---- GEMM1: h = relu(state @ Wb1[e] + bb1[e]) -> sH (bf16) ----
        {
            float bias[NT_BB];
            #pragma unroll
            for (int nt = 0; nt < NT_BB; ++nt)
                bias[nt] = bb1[e * H1 + nbase + nt * 16];
            v8f acc[2 * NT_BB];
            #pragma unroll
            for (int q = 0; q < 2 * NT_BB; ++q)
                #pragma unroll
                for (int r = 0; r < 8; ++r) acc[q][r] = 0.f;

            const bf16_t* bp = PREP ?
                W1T + ((size_t)e * H1 + nbase) * KPAD1 + kb : nullptr;
            #pragma unroll
            for (int t = 0; t < 2; ++t) {
                const bf16_t* a0p = &sA[nlo * KPAD1 + t * 32 + ka];
                const bf16_t* a1p = &sA[(16 + nlo) * KPAD1 + t * 32 + ka];
                const v16bf a0 = mk_frag(ld8(a0p), ld8(a0p + 16));
                const v16bf a1 = mk_frag(ld8(a1p), ld8(a1p + 16));
                #pragma unroll
                for (int nt = 0; nt < NT_BB; ++nt) {
                    v16bf b;
                    if constexpr (PREP) {
                        const bf16_t* bq = bp + nt * 16 * KPAD1 + t * 32;
                        b = mk_frag(ld8(bq), ld8(bq + 16));
                    } else {
                        const float* W1 = Wb1 + (size_t)e * IN_D * H1;
                        const int n = nbase + nt * 16;
                        #pragma unroll
                        for (int j = 0; j < 16; ++j) {
                            const int k = t * 32 + kb + j + (j & 8);
                            b[j] = (bf16_t)((k < IN_D) ? W1[k * H1 + n] : 0.f);
                        }
                    }
                    acc[nt]         = wmma_bf16(a0, b, acc[nt]);
                    acc[NT_BB + nt] = wmma_bf16(a1, b, acc[NT_BB + nt]);
                }
            }
            #pragma unroll
            for (int nt = 0; nt < NT_BB; ++nt) {
                const int n = nbase + nt * 16;
                #pragma unroll
                for (int r = 0; r < 8; ++r) {
                    float v0 = acc[nt][r] + bias[nt];
                    float v1 = acc[NT_BB + nt][r] + bias[nt];
                    v0 = v0 > 0.f ? v0 : 0.f;
                    v1 = v1 > 0.f ? v1 : 0.f;
                    sH[(mhi + r) * SSTR + n]      = (bf16_t)v0;
                    sH[(16 + mhi + r) * SSTR + n] = (bf16_t)v1;
                }
            }
        }
        __syncthreads();

        // per-row mixture weights for this expert
        float wr0[8], wr1[8];
        #pragma unroll
        for (int r = 0; r < 8; ++r) {
            wr0[r] = sW[(mhi + r) * NE + e];
            wr1[r] = sW[(16 + mhi + r) * NE + e];
        }

        // ---- GEMM2 + combine: sF += w[b,e] * relu(h @ Wb2[e] + bb2) ----
        {
            float bias[NT_BB];
            #pragma unroll
            for (int nt = 0; nt < NT_BB; ++nt)
                bias[nt] = bb2[e * H2 + nbase + nt * 16];
            v8f acc[2 * NT_BB];
            #pragma unroll
            for (int q = 0; q < 2 * NT_BB; ++q)
                #pragma unroll
                for (int r = 0; r < 8; ++r) acc[q][r] = 0.f;

            const bf16_t* bp = PREP ?
                W2T + ((size_t)e * H2 + nbase) * H1 + kb : nullptr;
            const float* wcol = PREP ? nullptr :
                Wb2 + (size_t)e * H1 * H2 + (size_t)kb * H2 + nbase;
            #pragma unroll 2
            for (int t = 0; t < 32; ++t) {
                const bf16_t* a0p = &sH[nlo * SSTR + t * 32 + ka];
                const bf16_t* a1p = &sH[(16 + nlo) * SSTR + t * 32 + ka];
                const v16bf a0 = mk_frag(ld8(a0p), ld8(a0p + 16));
                const v16bf a1 = mk_frag(ld8(a1p), ld8(a1p + 16));
                #pragma unroll
                for (int nt = 0; nt < NT_BB; ++nt) {
                    v16bf b;
                    if constexpr (PREP) {
                        const bf16_t* bq = bp + nt * 16 * H1 + t * 32;
                        b = mk_frag(ld8(bq), ld8(bq + 16));
                    } else {
                        #pragma unroll
                        for (int j = 0; j < 16; ++j)
                            b[j] = (bf16_t)wcol[(size_t)(t * 32 + j + (j & 8)) * H2 + nt * 16];
                    }
                    acc[nt]         = wmma_bf16(a0, b, acc[nt]);
                    acc[NT_BB + nt] = wmma_bf16(a1, b, acc[NT_BB + nt]);
                }
            }
            // RMW accumulate into LDS f buffer (each (m,n) owned by one lane)
            #pragma unroll
            for (int nt = 0; nt < NT_BB; ++nt) {
                const int n = nbase + nt * 16;
                #pragma unroll
                for (int r = 0; r < 8; ++r) {
                    float v0 = acc[nt][r] + bias[nt];
                    float v1 = acc[NT_BB + nt][r] + bias[nt];
                    v0 = v0 > 0.f ? v0 : 0.f;              // relu (2nd idempotent)
                    v1 = v1 > 0.f ? v1 : 0.f;
                    sF[(mhi + r) * FSTR + n]      += wr0[r] * v0;
                    sF[(16 + mhi + r) * FSTR + n] += wr1[r] * v1;
                }
            }
        }
        __syncthreads();   // all waves done reading sH before next expert
    }

    // ---- f = relu(sum_e w*feat) -> sH (bf16, buffer reuse) ----
    for (int nt = 0; nt < NT_BB; ++nt) {
        const int n = nbase + nt * 16;
        #pragma unroll
        for (int r = 0; r < 8; ++r) {
            float v0 = sF[(mhi + r) * FSTR + n];
            float v1 = sF[(16 + mhi + r) * FSTR + n];
            v0 = v0 > 0.f ? v0 : 0.f;
            v1 = v1 > 0.f ? v1 : 0.f;
            sH[(mhi + r) * SSTR + n]      = (bf16_t)v0;
            sH[(16 + mhi + r) * SSTR + n] = (bf16_t)v1;
        }
    }
    __syncthreads();

    // =================== per-context head layer 1 ===================
    for (int c = 0; c < NC; ++c) {
        if (!sPresent[c]) continue;                  // block-uniform skip
        float bias[NT_H];
        #pragma unroll
        for (int nt = 0; nt < NT_H; ++nt)
            bias[nt] = bh1[c * HHD + nbaseH + nt * 16];
        v8f acc[2 * NT_H];
        #pragma unroll
        for (int q = 0; q < 2 * NT_H; ++q)
            #pragma unroll
            for (int r = 0; r < 8; ++r) acc[q][r] = 0.f;

        const bf16_t* bp = PREP ?
            WhT + ((size_t)c * HHD + nbaseH) * H2 + kb : nullptr;
        const float* wcol = PREP ? nullptr :
            Wh1 + (size_t)c * H2 * HHD + (size_t)kb * HHD + nbaseH;
        #pragma unroll 2
        for (int t = 0; t < 32; ++t) {
            const bf16_t* a0p = &sH[nlo * SSTR + t * 32 + ka];
            const bf16_t* a1p = &sH[(16 + nlo) * SSTR + t * 32 + ka];
            const v16bf a0 = mk_frag(ld8(a0p), ld8(a0p + 16));
            const v16bf a1 = mk_frag(ld8(a1p), ld8(a1p + 16));
            #pragma unroll
            for (int nt = 0; nt < NT_H; ++nt) {
                v16bf b;
                if constexpr (PREP) {
                    const bf16_t* bq = bp + nt * 16 * H2 + t * 32;
                    b = mk_frag(ld8(bq), ld8(bq + 16));
                } else {
                    #pragma unroll
                    for (int j = 0; j < 16; ++j)
                        b[j] = (bf16_t)wcol[(size_t)(t * 32 + j + (j & 8)) * HHD + nt * 16];
                }
                acc[nt]        = wmma_bf16(a0, b, acc[nt]);
                acc[NT_H + nt] = wmma_bf16(a1, b, acc[NT_H + nt]);
            }
        }
        #pragma unroll
        for (int nt = 0; nt < NT_H; ++nt) {
            const int n = nbaseH + nt * 16;
            #pragma unroll
            for (int r = 0; r < 8; ++r) {
                const int m0 = mhi + r, m1 = 16 + mhi + r;
                if (sC[m0] == c) {
                    float v = acc[nt][r] + bias[nt];
                    sY[m0 * YSTR + n] = (bf16_t)(v > 0.f ? v : 0.f);
                }
                if (sC[m1] == c) {
                    float v = acc[NT_H + nt][r] + bias[nt];
                    sY[m1 * YSTR + n] = (bf16_t)(v > 0.f ? v : 0.f);
                }
            }
        }
    }
    __syncthreads();

    // =================== head layer 2 (N=4, VALU + wave reduce) ============
    #pragma unroll
    for (int rr = 0; rr < 2; ++rr) {
        const int m  = wave + rr * 16;               // rows 0..31
        const int cm = sC[m];
        const int o  = lane & 3;
        const int ch = lane >> 2;                    // 8 chunks of 64 j's
        const float* Wo = Wh2 + (size_t)cm * HHD * OUTD;
        float sum = 0.f;
        const int j0 = ch * 64;
        for (int j = j0; j < j0 + 64; ++j)
            sum += (float)sY[m * YSTR + j] * Wo[j * OUTD + o];
        sum += __shfl_xor(sum, 4);
        sum += __shfl_xor(sum, 8);
        sum += __shfl_xor(sum, 16);
        if (lane < 4)
            out[(size_t)(b0 + m) * OUTD + o] = sum + bh2[cm * OUTD + o];
    }
}

extern "C" void kernel_launch(void* const* d_in, const int* in_sizes, int n_in,
                              void* d_out, int out_size, void* d_ws, size_t ws_size,
                              hipStream_t stream) {
    (void)n_in; (void)out_size;
    const float* state = (const float*)d_in[0];
    const int*   cidx  = (const int*)  d_in[1];
    const float* Wte   = (const float*)d_in[2];
    const float* Wb1   = (const float*)d_in[3];
    const float* bb1   = (const float*)d_in[4];
    const float* Wb2   = (const float*)d_in[5];
    const float* bb2   = (const float*)d_in[6];
    const float* Wh1   = (const float*)d_in[7];
    const float* bh1   = (const float*)d_in[8];
    const float* Wh2   = (const float*)d_in[9];
    const float* bh2   = (const float*)d_in[10];
    float* out = (float*)d_out;

    const int B    = in_sizes[1];        // batch = element count of c
    const int grid = B / BT;

    const size_t need =
        (W1T_ELEMS + W2T_ELEMS + WHT_ELEMS) * sizeof(bf16_t);

    if (ws_size >= need) {
        bf16_t* W1T = (bf16_t*)d_ws;
        bf16_t* W2T = W1T + W1T_ELEMS;
        bf16_t* WhT = W2T + W2T_ELEMS;
        // weight transpose+convert (ordered on stream before main kernel)
        transpose_cvt<<<dim3(NE, KPAD1 / 32, H1 / 32), 256, 0, stream>>>(
            Wb1, W1T, IN_D, H1, KPAD1);
        transpose_cvt<<<dim3(NE, H1 / 32, H2 / 32), 256, 0, stream>>>(
            Wb2, W2T, H1, H2, H1);
        transpose_cvt<<<dim3(NC, H2 / 32, HHD / 32), 256, 0, stream>>>(
            Wh1, WhT, H2, HHD, H2);
        fused_moe_actor<true><<<grid, THREADS, 0, stream>>>(
            state, cidx, Wte, Wb1, bb1, Wb2, bb2, Wh1, bh1, Wh2, bh2,
            W1T, W2T, WhT, out);
    } else {
        fused_moe_actor<false><<<grid, THREADS, 0, stream>>>(
            state, cidx, Wte, Wb1, bb1, Wb2, bb2, Wh1, bh1, Wh2, bh2,
            nullptr, nullptr, nullptr, out);
    }
}
